// HeteroRGCN_74045236183059
// MI455X (gfx1250) — matrix operand order repaired and checked by reading, not compiled, not run on hardware
//
#include <hip/hip_runtime.h>
#include <hip/hip_bf16.h>

#define N_ART  50000
#define N_ENT  100000
#define N_FACT 50000
#define NEDGE  200000
#define FEAT   769
#define KP1    800      // 769 padded to a multiple of 32
#define HID    128
#define LDSP   40       // padded LDS row stride in bf16 (80 B -> bank-conflict free)

typedef __attribute__((ext_vector_type(16))) __bf16 bf16x16;
typedef __attribute__((ext_vector_type(8)))  __bf16 bf16x8;
typedef __attribute__((ext_vector_type(8)))  float  f32x8;

// CDNA5 async global->LDS path (ASYNCcnt), with sync fallback if the
// toolchain doesn't declare the builtins.
#if defined(__gfx1250__) && \
    __has_builtin(__builtin_amdgcn_global_load_async_to_lds_b128) && \
    __has_builtin(__builtin_amdgcn_s_wait_asynccnt)
#define USE_ASYNC_LDS 1
typedef int v4i __attribute__((vector_size(16)));
typedef v4i __attribute__((address_space(1))) ga_v4i;   // global int4
typedef v4i __attribute__((address_space(3))) ls_v4i;   // LDS int4
#else
#define USE_ASYNC_LDS 0
#endif

// ---------------------------------------------------------------------------
// fp32 -> bf16 cast with K padding and optional ReLU.
// src: [R][K] fp32 row-major, dst: [R][Kp] bf16 row-major (pad zero-filled)
// ---------------------------------------------------------------------------
__global__ void cast_pad_k(const float* __restrict__ src, __bf16* __restrict__ dst,
                           int R, int K, int Kp, int do_relu) {
  size_t i = (size_t)blockIdx.x * blockDim.x + threadIdx.x;
  size_t total = (size_t)R * Kp;
  if (i >= total) return;
  int k = (int)(i % Kp);
  size_t r = i / Kp;
  float v = 0.0f;
  if (k < K) {
    v = src[r * (size_t)K + k];
    if (do_relu) v = fmaxf(v, 0.0f);
  }
  dst[i] = (__bf16)v;
}

// ---------------------------------------------------------------------------
// C[M][128] = A[M][Kp](bf16) @ B[128][Kp](bf16)^T (+ bias), fp32 out.
// 256 threads = 8 waves. Block tile 64x128. Wave (w): mSub = w&3 (16 rows),
// nGrp = w>>2 (64 cols = 4 x 16-col WMMA subtiles).
// Double-buffered LDS tiles for A (64x32) and B (128x32); next chunk is
// fetched with GLOBAL_LOAD_ASYNC_TO_LDS_B128 while WMMAs run on the current
// buffer; producer/consumer sync = s_wait_asynccnt + workgroup barrier.
// v_wmma_f32_16x16x32_bf16 fragment layout per ISA 7.12.2:
//   A frag: lane<16 -> K = {0..7,16..23}+k0 ; lane>=16 -> K = {8..15,24..31}+k0
//   B frag: N = lane&15, K same as A frag: lane<16 -> {0..7,16..23} ; lane>=16 -> {8..15,24..31}
//   C/D   : N = lane&15, VGPR r -> M = r + 8*(lane>>4)
// ---------------------------------------------------------------------------
__global__ __launch_bounds__(256)
void gemm_bf16_wmma_k(const __bf16* __restrict__ A, const __bf16* __restrict__ B,
                      const float* __restrict__ bias, float* __restrict__ C,
                      int M, int Kp) {
  __shared__ __align__(16) __bf16 As[2][64][LDSP];    // 10 KB
  __shared__ __align__(16) __bf16 Bs[2][128][LDSP];   // 20 KB

  const int tid  = threadIdx.x;
  const int lane = tid & 31;
  const int wave = tid >> 5;
  const int mSub = wave & 3;
  const int nGrp = wave >> 2;
  const int m0   = blockIdx.x * 64;
  const int half = lane >> 4;       // 0 | 1
  const int l16  = lane & 15;

  // staging assignment: each thread moves three 16 B segments per chunk:
  //   A row (0..63) seg (0..3), B rows (0..63)+(64..127) same seg
  const int arow = tid >> 2;
  const int aseg = tid & 3;

  // issue one chunk's tile loads into buffer b
  auto stage = [&](int b, int kc) {
    const int k0 = kc << 5;
    int gr = m0 + arow;
    gr = (gr < M) ? gr : (M - 1);   // clamp: garbage rows never stored
    const __bf16* gA  = A + (size_t)gr * Kp + k0 + aseg * 8;
    const __bf16* gB0 = B + (size_t)arow * Kp + k0 + aseg * 8;
    const __bf16* gB1 = B + (size_t)(arow + 64) * Kp + k0 + aseg * 8;
    __bf16* lA  = &As[b][arow][aseg * 8];
    __bf16* lB0 = &Bs[b][arow][aseg * 8];
    __bf16* lB1 = &Bs[b][arow + 64][aseg * 8];
#if USE_ASYNC_LDS
    __builtin_amdgcn_global_load_async_to_lds_b128((ga_v4i*)gA,  (ls_v4i*)lA,  0, 0);
    __builtin_amdgcn_global_load_async_to_lds_b128((ga_v4i*)gB0, (ls_v4i*)lB0, 0, 0);
    __builtin_amdgcn_global_load_async_to_lds_b128((ga_v4i*)gB1, (ls_v4i*)lB1, 0, 0);
#else
    *(bf16x8*)lA  = *(const bf16x8*)gA;
    *(bf16x8*)lB0 = *(const bf16x8*)gB0;
    *(bf16x8*)lB1 = *(const bf16x8*)gB1;
#endif
  };

  f32x8 acc[4];
  #pragma unroll
  for (int s = 0; s < 4; ++s)
    #pragma unroll
    for (int r = 0; r < 8; ++r) acc[s][r] = 0.0f;

  const int nK = Kp >> 5;
  stage(0, 0);
  int buf = 0;
  for (int kc = 0; kc < nK; ++kc) {
    if (kc + 1 < nK) {
      stage(buf ^ 1, kc + 1);        // prefetch next chunk asynchronously
#if USE_ASYNC_LDS
      __builtin_amdgcn_s_wait_asynccnt(3);   // current chunk's 3 loads done
#endif
    } else {
#if USE_ASYNC_LDS
      __builtin_amdgcn_s_wait_asynccnt(0);
#endif
    }
    __syncthreads();                 // cross-wave visibility of LDS tile

    // --- A fragment for this wave's 16-row stripe ---
    bf16x16 af;
    {
      const int r  = mSub * 16 + l16;
      const int kb = half * 8;
      bf16x8 lo = *(const bf16x8*)(&As[buf][r][kb]);
      bf16x8 hi = *(const bf16x8*)(&As[buf][r][kb + 16]);
      #pragma unroll
      for (int i = 0; i < 8; ++i) { af[i] = lo[i]; af[i + 8] = hi[i]; }
    }

    // --- 4 B fragments from LDS + WMMA ---
    #pragma unroll
    for (int s = 0; s < 4; ++s) {
      const int n = nGrp * 64 + s * 16 + l16;
      const __bf16* bp = &Bs[buf][n][half * 8];
      bf16x8 blo = *(const bf16x8*)(bp);
      bf16x8 bhi = *(const bf16x8*)(bp + 16);
      bf16x16 bfr;
      #pragma unroll
      for (int i = 0; i < 8; ++i) { bfr[i] = blo[i]; bfr[i + 8] = bhi[i]; }
      acc[s] = __builtin_amdgcn_wmma_f32_16x16x32_bf16(
          false, af, false, bfr, (short)0, acc[s], false, false);
    }
    __syncthreads();                 // done reading buf before it is re-staged
    buf ^= 1;
  }

  // --- epilogue: bias + store ---
  #pragma unroll
  for (int s = 0; s < 4; ++s) {
    const int n  = nGrp * 64 + s * 16 + l16;
    const float bv = bias ? bias[n] : 0.0f;
    const int rbase = m0 + mSub * 16 + half * 8;
    #pragma unroll
    for (int r = 0; r < 8; ++r) {
      const int row = rbase + r;
      if (row < M) C[(size_t)row * HID + n] = acc[s][r] + bv;
    }
  }
}

// ---------------------------------------------------------------------------
// degree count per destination node
// ---------------------------------------------------------------------------
__global__ void count_k(const int* __restrict__ di, unsigned* __restrict__ cnt, int ne) {
  int e = blockIdx.x * blockDim.x + threadIdx.x;
  if (e < ne) atomicAdd(&cnt[di[e]], 1u);
}

// ---------------------------------------------------------------------------
// agg[di] += y[si] over edges; 32 threads per edge, 4 floats per thread
// ---------------------------------------------------------------------------
__global__ void scatter_k(const float* __restrict__ y, const int* __restrict__ si,
                          const int* __restrict__ di, float* __restrict__ agg, int ne) {
  size_t idx = (size_t)blockIdx.x * blockDim.x + threadIdx.x;
  size_t total = (size_t)ne * 32;
  if (idx >= total) return;
  const int e = (int)(idx >> 5);
  const int c = (int)(idx & 31) * 4;
  const float4 v = *(const float4*)(y + (size_t)si[e] * HID + c);
  float* p = agg + (size_t)di[e] * HID + c;
  __hip_atomic_fetch_add(p + 0, v.x, __ATOMIC_RELAXED, __HIP_MEMORY_SCOPE_AGENT);
  __hip_atomic_fetch_add(p + 1, v.y, __ATOMIC_RELAXED, __HIP_MEMORY_SCOPE_AGENT);
  __hip_atomic_fetch_add(p + 2, v.z, __ATOMIC_RELAXED, __HIP_MEMORY_SCOPE_AGENT);
  __hip_atomic_fetch_add(p + 3, v.w, __ATOMIC_RELAXED, __HIP_MEMORY_SCOPE_AGENT);
}

// ---------------------------------------------------------------------------
// t = lin_l + agg/max(cnt,1); L2-normalize row; out += t/max(||t||, 1e-12)
// one 128-thread block per destination row
// ---------------------------------------------------------------------------
__global__ __launch_bounds__(128)
void combine_k(const float* __restrict__ lin, const float* __restrict__ agg,
               const unsigned* __restrict__ cnt, float* __restrict__ out) {
  const int row = blockIdx.x;
  const int h = threadIdx.x;
  __shared__ float red[128];
  const float inv = 1.0f / fmaxf((float)cnt[row], 1.0f);
  const float t = lin[(size_t)row * HID + h] + agg[(size_t)row * HID + h] * inv;
  red[h] = t * t;
  __syncthreads();
  for (int s = 64; s > 0; s >>= 1) {
    if (h < s) red[h] += red[h + s];
    __syncthreads();
  }
  const float rn = 1.0f / fmaxf(sqrtf(red[0]), 1e-12f);
  out[(size_t)row * HID + h] += t * rn;
}

__global__ void relu_k(float* __restrict__ x, size_t n) {
  size_t i = (size_t)blockIdx.x * blockDim.x + threadIdx.x;
  if (i < n) x[i] = fmaxf(x[i], 0.0f);
}

// ---------------------------------------------------------------------------
// head: z = relu(h @ W1^T + b1) [64], logits = z @ W2^T + b2 [2]
// one 64-thread block per article row (tiny: ~0.8 GFLOP total)
// ---------------------------------------------------------------------------
__global__ __launch_bounds__(64)
void head_k(const float* __restrict__ h, const float* __restrict__ w1,
            const float* __restrict__ b1, const float* __restrict__ w2,
            const float* __restrict__ b2, float* __restrict__ logits) {
  const int row = blockIdx.x;
  const int j = threadIdx.x;
  __shared__ float hs[128];
  __shared__ float zs[64];
  hs[j]      = h[(size_t)row * HID + j];
  hs[j + 64] = h[(size_t)row * HID + j + 64];
  __syncthreads();
  float acc = b1[j];
  #pragma unroll 8
  for (int k = 0; k < 128; ++k) acc += hs[k] * w1[j * 128 + k];
  zs[j] = fmaxf(acc, 0.0f);
  __syncthreads();
  if (j < 2) {
    float a = b2[j];
    #pragma unroll 8
    for (int k = 0; k < 64; ++k) a += zs[k] * w2[j * 64 + k];
    logits[(size_t)row * 2 + j] = a;
  }
}

// ---------------------------------------------------------------------------
static inline unsigned ceil_div_u(size_t a, size_t b) { return (unsigned)((a + b - 1) / b); }

extern "C" void kernel_launch(void* const* d_in, const int* in_sizes, int n_in,
                              void* d_out, int out_size, void* d_ws, size_t ws_size,
                              hipStream_t stream) {
  (void)in_sizes; (void)n_in; (void)out_size; (void)ws_size;

  // ---- inputs (setup_inputs order) ----
  const float* x_art = (const float*)d_in[0];
  const float* x_ent = (const float*)d_in[1];
  const float* x_fct = (const float*)d_in[2];
  const int* ei[5] = { (const int*)d_in[3], (const int*)d_in[4], (const int*)d_in[5],
                       (const int*)d_in[6], (const int*)d_in[7] };
  const float* c1_ll_w = (const float*)d_in[8];
  const float* c1_ll_b = (const float*)d_in[9];
  const float* c1_lr_w = (const float*)d_in[10];
  const float* c2_ll_w = (const float*)d_in[11];
  const float* c2_ll_b = (const float*)d_in[12];
  const float* c2_lr_w = (const float*)d_in[13];
  const float* head_w1 = (const float*)d_in[14];
  const float* head_b1 = (const float*)d_in[15];
  const float* head_w2 = (const float*)d_in[16];
  const float* head_b2 = (const float*)d_in[17];

  float* out = (float*)d_out;

  // ---- workspace layout ----
  char* ws = (char*)d_ws;
  size_t off = 0;
  auto carve = [&](size_t bytes) -> char* {
    char* p = ws + off;
    off += (bytes + 255) & ~(size_t)255;
    return p;
  };
  __bf16* xb[3];
  xb[0] = (__bf16*)carve((size_t)N_ART  * KP1 * 2);
  xb[1] = (__bf16*)carve((size_t)N_ENT  * KP1 * 2);
  xb[2] = (__bf16*)carve((size_t)N_FACT * KP1 * 2);
  __bf16* w1l = (__bf16*)carve((size_t)5 * HID * KP1 * 2);
  __bf16* w1r = (__bf16*)carve((size_t)5 * HID * KP1 * 2);
  __bf16* w2l = (__bf16*)carve((size_t)5 * HID * HID * 2);
  __bf16* w2r = (__bf16*)carve((size_t)5 * HID * HID * 2);
  float* tmpL = (float*)carve((size_t)N_ENT * HID * 4);
  float* tmpY = (float*)carve((size_t)N_ENT * HID * 4);
  float* agg  = (float*)carve((size_t)N_ENT * HID * 4);
  unsigned* cnt = (unsigned*)carve((size_t)N_ENT * 4);
  float* h1[3];
  h1[0] = (float*)carve((size_t)N_ART  * HID * 4);
  h1[1] = (float*)carve((size_t)N_ENT  * HID * 4);
  h1[2] = (float*)carve((size_t)N_FACT * HID * 4);
  __bf16* hb[3];
  hb[0] = (__bf16*)carve((size_t)N_ART  * HID * 2);
  hb[1] = (__bf16*)carve((size_t)N_ENT  * HID * 2);
  hb[2] = (__bf16*)carve((size_t)N_FACT * HID * 2);

  const int Ns[3] = { N_ART, N_ENT, N_FACT };
  // edge type i: (src_type, dst_type) indices into {art=0, ent=1, fact=2}
  const int esrc[5] = { 0, 0, 1, 1, 2 };
  const int edst[5] = { 1, 2, 0, 2, 1 };

  // output regions: logits | h2_art | h2_ent | h2_fact
  float* logits = out;
  float* h2[3];
  h2[0] = out + (size_t)N_ART * 2;
  h2[1] = h2[0] + (size_t)N_ART * HID;
  h2[2] = h2[1] + (size_t)N_ENT * HID;
  const size_t h2_total = (size_t)(N_ART + N_ENT + N_FACT) * HID;

  // ---- phase 0: casts + zero-init ----
  {
    const size_t tot0 = (size_t)N_ART * KP1;
    cast_pad_k<<<ceil_div_u(tot0, 256), 256, 0, stream>>>(x_art, xb[0], N_ART, FEAT, KP1, 0);
    const size_t tot1 = (size_t)N_ENT * KP1;
    cast_pad_k<<<ceil_div_u(tot1, 256), 256, 0, stream>>>(x_ent, xb[1], N_ENT, FEAT, KP1, 0);
    const size_t tot2 = (size_t)N_FACT * KP1;
    cast_pad_k<<<ceil_div_u(tot2, 256), 256, 0, stream>>>(x_fct, xb[2], N_FACT, FEAT, KP1, 0);
    const size_t wtot = (size_t)5 * HID * KP1;
    cast_pad_k<<<ceil_div_u(wtot, 256), 256, 0, stream>>>(c1_ll_w, w1l, 5 * HID, FEAT, KP1, 0);
    cast_pad_k<<<ceil_div_u(wtot, 256), 256, 0, stream>>>(c1_lr_w, w1r, 5 * HID, FEAT, KP1, 0);
    const size_t wtot2 = (size_t)5 * HID * HID;
    cast_pad_k<<<ceil_div_u(wtot2, 256), 256, 0, stream>>>(c2_ll_w, w2l, 5 * HID, HID, HID, 0);
    cast_pad_k<<<ceil_div_u(wtot2, 256), 256, 0, stream>>>(c2_lr_w, w2r, 5 * HID, HID, HID, 0);
  }
  (void)hipMemsetAsync(h1[0], 0, (size_t)N_ART  * HID * 4, stream);
  (void)hipMemsetAsync(h1[1], 0, (size_t)N_ENT  * HID * 4, stream);
  (void)hipMemsetAsync(h1[2], 0, (size_t)N_FACT * HID * 4, stream);
  (void)hipMemsetAsync(d_out, 0, ((size_t)N_ART * 2 + h2_total) * 4, stream);

  // ---- layer 1 ----
  for (int i = 0; i < 5; ++i) {
    const int s = esrc[i], d = edst[i];
    const int Md = Ns[d], Ms = Ns[s];
    gemm_bf16_wmma_k<<<ceil_div_u(Md, 64), 256, 0, stream>>>(
        xb[d], w1l + (size_t)i * HID * KP1, c1_ll_b + (size_t)i * HID, tmpL, Md, KP1);
    gemm_bf16_wmma_k<<<ceil_div_u(Ms, 64), 256, 0, stream>>>(
        xb[s], w1r + (size_t)i * HID * KP1, nullptr, tmpY, Ms, KP1);
    (void)hipMemsetAsync(agg, 0, (size_t)Md * HID * 4, stream);
    (void)hipMemsetAsync(cnt, 0, (size_t)Md * 4, stream);
    count_k<<<ceil_div_u(NEDGE, 256), 256, 0, stream>>>(ei[i] + NEDGE, cnt, NEDGE);
    scatter_k<<<ceil_div_u((size_t)NEDGE * 32, 256), 256, 0, stream>>>(
        tmpY, ei[i], ei[i] + NEDGE, agg, NEDGE);
    combine_k<<<Md, 128, 0, stream>>>(tmpL, agg, cnt, h1[d]);
  }

  // ---- relu + cast to bf16 for layer 2 ----
  for (int t = 0; t < 3; ++t) {
    const size_t tot = (size_t)Ns[t] * HID;
    cast_pad_k<<<ceil_div_u(tot, 256), 256, 0, stream>>>(h1[t], hb[t], Ns[t], HID, HID, 1);
  }

  // ---- layer 2 (accumulates into d_out h2 regions) ----
  for (int i = 0; i < 5; ++i) {
    const int s = esrc[i], d = edst[i];
    const int Md = Ns[d], Ms = Ns[s];
    gemm_bf16_wmma_k<<<ceil_div_u(Md, 64), 256, 0, stream>>>(
        hb[d], w2l + (size_t)i * HID * HID, c2_ll_b + (size_t)i * HID, tmpL, Md, HID);
    gemm_bf16_wmma_k<<<ceil_div_u(Ms, 64), 256, 0, stream>>>(
        hb[s], w2r + (size_t)i * HID * HID, nullptr, tmpY, Ms, HID);
    (void)hipMemsetAsync(agg, 0, (size_t)Md * HID * 4, stream);
    (void)hipMemsetAsync(cnt, 0, (size_t)Md * 4, stream);
    count_k<<<ceil_div_u(NEDGE, 256), 256, 0, stream>>>(ei[i] + NEDGE, cnt, NEDGE);
    scatter_k<<<ceil_div_u((size_t)NEDGE * 32, 256), 256, 0, stream>>>(
        tmpY, ei[i], ei[i] + NEDGE, agg, NEDGE);
    combine_k<<<Md, 128, 0, stream>>>(tmpL, agg, cnt, h2[d]);
  }

  // ---- final relu on h2 outputs (in place in d_out) ----
  relu_k<<<ceil_div_u(h2_total, 256), 256, 0, stream>>>(h2[0], h2_total);

  // ---- head MLP on articles ----
  head_k<<<N_ART, 64, 0, stream>>>(h2[0], head_w1, head_b1, head_w2, head_b2, logits);
}
